// GraphAttentionNetworks_23330262352556
// MI455X (gfx1250) — matrix operand order repaired and checked by reading, not compiled, not run on hardware
//
#include <hip/hip_runtime.h>
#include <hip/hip_bf16.h>

typedef __attribute__((ext_vector_type(16))) __bf16 v16bf;
typedef __attribute__((ext_vector_type(8)))  __bf16 v8bf;
typedef __attribute__((ext_vector_type(8)))  float  v8f;
typedef __attribute__((ext_vector_type(4)))  float  v4f;

#define SEQ 1024
#define FD  128
#define BPN 64

__device__ __forceinline__ v16bf cat8(v8bf lo, v8bf hi) {
    return __builtin_shufflevector(lo, hi, 0,1,2,3,4,5,6,7,8,9,10,11,12,13,14,15);
}

__device__ __forceinline__ v8bf pack8(v4f a, v4f b) {
    v8bf r;
    r[0]=(__bf16)a[0]; r[1]=(__bf16)a[1]; r[2]=(__bf16)a[2]; r[3]=(__bf16)a[3];
    r[4]=(__bf16)b[0]; r[5]=(__bf16)b[1]; r[6]=(__bf16)b[2]; r[7]=(__bf16)b[3];
    return r;
}

// Regular-temporal load+convert (W: reused by all 1536 blocks -> keep cached)
__device__ __forceinline__ v8bf cvt8(const float* p) {
    return pack8(*(const v4f*)p, *(const v4f*)(p + 4));
}
// Non-temporal load+convert (q/k/v inputs: streamed exactly once)
__device__ __forceinline__ v8bf cvt8_nt(const float* p) {
    v4f a = __builtin_nontemporal_load((const v4f*)p);
    v4f b = __builtin_nontemporal_load((const v4f*)(p + 4));
    return pack8(a, b);
}

// ---------------------------------------------------------------------------
// Stage 1: project {query,key,value} through W (x @ W^T), f32 -> bf16.
// Q,K row-major (s,f); V transposed (f,s) so stage-2 B-operand loads are
// contiguous along K.
// grid = 3 * 65536/128 = 1536 blocks, 256 threads (8 waves, 16 rows/wave).
// ---------------------------------------------------------------------------
__global__ __launch_bounds__(256)
void proj_kernel(const float* __restrict__ q, const float* __restrict__ k,
                 const float* __restrict__ v, const float* __restrict__ W,
                 __hip_bfloat16* __restrict__ Qp, __hip_bfloat16* __restrict__ Kp,
                 __hip_bfloat16* __restrict__ Vtp)
{
    __bf16* Qw = reinterpret_cast<__bf16*>(Qp);
    __bf16* Kw = reinterpret_cast<__bf16*>(Kp);
    __bf16* Vt = reinterpret_cast<__bf16*>(Vtp);

    const int lane = threadIdx.x & 31;
    const int wave = threadIdx.x >> 5;
    const int half = lane >> 4;
    const int l    = lane & 15;
    const int seg  = blockIdx.x / 512;                     // 0=q,1=k,2=v
    const int rows = (blockIdx.x % 512) * 128 + wave * 16; // row in segment

    const float* src = (seg == 0) ? q : (seg == 1) ? k : v;

    // A fragments: 16 rows x 128 K of input, 4 K-steps of 32 (streamed once)
    v16bf aF[4];
    {
        const float* rp = src + (size_t)(rows + l) * FD;
        #pragma unroll
        for (int ks = 0; ks < 4; ++ks) {
            const int f0 = ks * 32 + half * 8;
            aF[ks] = cat8(cvt8_nt(rp + f0), cvt8_nt(rp + f0 + 16));
        }
    }
    #pragma unroll
    for (int nt = 0; nt < 8; ++nt) {
        v8f acc = {};
        const float* wrow = W + (size_t)(nt * 16 + l) * FD;  // B[k=f][n=g]=W[g,f]
        #pragma unroll
        for (int ks = 0; ks < 4; ++ks) {
            const int f0 = ks * 32 + half * 8;
            v16bf bF = cat8(cvt8(wrow + f0), cvt8(wrow + f0 + 16));
            acc = __builtin_amdgcn_wmma_f32_16x16x32_bf16(
                false, aF[ks], false, bF, (short)0, acc, false, false);
        }
        const int g = nt * 16 + l;                         // N = output feature
        if (seg < 2) {
            __bf16* dst = (seg == 0) ? Qw : Kw;
            #pragma unroll
            for (int j = 0; j < 8; ++j) {
                const int s = rows + j + 8 * half;         // M
                dst[(size_t)s * FD + g] = (__bf16)acc[j];
            }
        } else {
            // V transposed: Vt[bp][f][s], 8 consecutive s per lane -> one b128
            const int sBase = rows + 8 * half;
            const int bp = sBase >> 10;
            const int sI = sBase & 1023;
            v8bf pk;
            #pragma unroll
            for (int j = 0; j < 8; ++j) pk[j] = (__bf16)acc[j];
            *(v8bf*)(Vt + ((size_t)bp * FD + g) * SEQ + sI) = pk;
        }
    }
}

// ---------------------------------------------------------------------------
// Stage 2: per (bp, 16-query tile): scores -> leaky -> softmax -> attn out,
// then out = leaky(attn @ V). grid = 64*64 = 4096 blocks, 256 threads.
// Wave w owns keys [w*128, w*128+128) in phase 1/2 and output feature tile
// [w*16, w*16+16) in phase 3. attn/out stores are non-temporal so the 268MB
// attn stream does not evict the 48MB (64x-reused) bf16 workspace from L2.
// ---------------------------------------------------------------------------
__global__ __launch_bounds__(256)
void attn_kernel(const __hip_bfloat16* __restrict__ Qp,
                 const __hip_bfloat16* __restrict__ Kp,
                 const __hip_bfloat16* __restrict__ Vtp,
                 float* __restrict__ outP, float* __restrict__ attnP)
{
    const __bf16* Qw = reinterpret_cast<const __bf16*>(Qp);
    const __bf16* Kw = reinterpret_cast<const __bf16*>(Kp);
    const __bf16* Vt = reinterpret_cast<const __bf16*>(Vtp);

    __shared__ __bf16 pbuf[16 * 1024];   // unnormalized softmax numerators (bf16)
    __shared__ float  wred[8 * 16];      // per-wave row max / row sum

    const int lane  = threadIdx.x & 31;
    const int wave  = threadIdx.x >> 5;
    const int half  = lane >> 4;
    const int l     = lane & 15;
    const int bp    = blockIdx.x >> 6;
    const int qBase = (blockIdx.x & 63) * 16;

    // Q A-fragments (16 rows x 128)
    v16bf aF[4];
    {
        const __bf16* qr = Qw + (size_t)(bp * SEQ + qBase + l) * FD;
        #pragma unroll
        for (int ks = 0; ks < 4; ++ks) {
            const int f0 = ks * 32 + half * 8;
            aF[ks] = cat8(*(const v8bf*)(qr + f0), *(const v8bf*)(qr + f0 + 16));
        }
    }

    // ---- phase 1: e = leaky(QK^T / sqrt(F)) for this wave's 128 keys
    const float invScale = 0.08838834764831845f;  // 1/sqrt(128)
    v8f  sc[8];
    float m[8];
    #pragma unroll
    for (int j = 0; j < 8; ++j) m[j] = -3.0e38f;

    const __bf16* kbase = Kw + (size_t)(bp * SEQ + wave * 128 + l) * FD;
    #pragma unroll
    for (int t = 0; t < 8; ++t) {
        if (t < 7)  // prefetch next K tile row for this lane
            __builtin_prefetch(kbase + (size_t)(t + 1) * 16 * FD, 0, 3);
        v8f acc = {};
        const __bf16* kr = kbase + (size_t)t * 16 * FD;
        #pragma unroll
        for (int ks = 0; ks < 4; ++ks) {
            const int f0 = ks * 32 + half * 8;
            v16bf bF = cat8(*(const v8bf*)(kr + f0), *(const v8bf*)(kr + f0 + 16));
            acc = __builtin_amdgcn_wmma_f32_16x16x32_bf16(
                false, aF[ks], false, bF, (short)0, acc, false, false);
        }
        #pragma unroll
        for (int j = 0; j < 8; ++j) {
            float a = acc[j] * invScale;
            float e = (a >= 0.f) ? a : 0.01f * a;
            acc[j] = e;
            m[j] = fmaxf(m[j], e);
        }
        sc[t] = acc;
    }
    // row max: butterfly inside each 16-lane half (rows j / j+8 live in halves)
    #pragma unroll
    for (int off = 1; off < 16; off <<= 1)
        #pragma unroll
        for (int j = 0; j < 8; ++j) m[j] = fmaxf(m[j], __shfl_xor(m[j], off, 32));
    if (l == 0)
        #pragma unroll
        for (int j = 0; j < 8; ++j) wred[wave * 16 + half * 8 + j] = m[j];
    __syncthreads();
    float rm[8];
    #pragma unroll
    for (int j = 0; j < 8; ++j) {
        float v = wred[half * 8 + j];
        #pragma unroll
        for (int w = 1; w < 8; ++w) v = fmaxf(v, wred[w * 16 + half * 8 + j]);
        rm[j] = v;
    }
    __syncthreads();  // wred reused for sums below

    // ---- phase 2: p = exp(e - max), stash bf16 p in LDS, reduce row sums
    float s[8];
    #pragma unroll
    for (int j = 0; j < 8; ++j) s[j] = 0.f;
    #pragma unroll
    for (int t = 0; t < 8; ++t) {
        #pragma unroll
        for (int j = 0; j < 8; ++j) {
            float p = __expf(sc[t][j] - rm[j]);
            sc[t][j] = p;
            s[j] += p;
            const int row = j + 8 * half;
            const int key = wave * 128 + t * 16 + l;
            pbuf[row * 1024 + key] = (__bf16)p;
        }
    }
    #pragma unroll
    for (int off = 1; off < 16; off <<= 1)
        #pragma unroll
        for (int j = 0; j < 8; ++j) s[j] += __shfl_xor(s[j], off, 32);
    if (l == 0)
        #pragma unroll
        for (int j = 0; j < 8; ++j) wred[wave * 16 + half * 8 + j] = s[j];
    __syncthreads();  // also orders pbuf stores before phase-3 LDS reads
    float inv[8];
    #pragma unroll
    for (int j = 0; j < 8; ++j) {
        float v = 0.f;
        #pragma unroll
        for (int w = 0; w < 8; ++w) v += wred[w * 16 + half * 8 + j];
        inv[j] = 1.0f / v;
    }

    // write normalized attn: f32, coalesced 64B/row-group, non-temporal
    float* ar = attnP + ((size_t)bp * SEQ + qBase) * SEQ;
    #pragma unroll
    for (int t = 0; t < 8; ++t) {
        const int key = wave * 128 + t * 16 + l;
        #pragma unroll
        for (int j = 0; j < 8; ++j) {
            const int row = j + 8 * half;
            float a = sc[t][j] * inv[j];
            __builtin_nontemporal_store(a, &ar[(size_t)row * SEQ + key]);
        }
    }

    // ---- phase 3: out tile cols [wave*16, +16) = leaky((p @ V) / rowsum)
    v8f acc = {};
    const __bf16* vtb = Vt + ((size_t)bp * FD + wave * 16 + l) * SEQ;
    for (int ks = 0; ks < 32; ++ks) {
        const int k0 = ks * 32;
        const int fa = k0 + half * 8;
        v16bf aP = cat8(*(const v8bf*)(pbuf + l * 1024 + fa),
                        *(const v8bf*)(pbuf + l * 1024 + fa + 16));
        const int fb = k0 + half * 8;
        v16bf bP = cat8(*(const v8bf*)(vtb + fb), *(const v8bf*)(vtb + fb + 16));
        acc = __builtin_amdgcn_wmma_f32_16x16x32_bf16(
            false, aP, false, bP, (short)0, acc, false, false);
    }
    float* orow = outP + ((size_t)(bp * SEQ + qBase)) * FD + wave * 16 + l;
    #pragma unroll
    for (int j = 0; j < 8; ++j) {
        const int row = j + 8 * half;
        float o = acc[j] * inv[j];
        o = (o >= 0.f) ? o : 0.01f * o;
        __builtin_nontemporal_store(o, &orow[(size_t)row * FD]);
    }
}

// ---------------------------------------------------------------------------
extern "C" void kernel_launch(void* const* d_in, const int* in_sizes, int n_in,
                              void* d_out, int out_size, void* d_ws, size_t ws_size,
                              hipStream_t stream) {
    const float* q = (const float*)d_in[0];
    const float* k = (const float*)d_in[1];
    const float* v = (const float*)d_in[2];
    const float* W = (const float*)d_in[3];

    float* outP  = (float*)d_out;
    float* attnP = outP + (size_t)BPN * SEQ * FD;   // out first, then attn

    __hip_bfloat16* Qw = (__hip_bfloat16*)d_ws;
    __hip_bfloat16* Kw = Qw + (size_t)BPN * SEQ * FD;
    __hip_bfloat16* Vt = Kw + (size_t)BPN * SEQ * FD;

    proj_kernel<<<1536, 256, 0, stream>>>(q, k, v, W, Qw, Kw, Vt);
    attn_kernel<<<4096, 256, 0, stream>>>(Qw, Kw, Vt, outP, attnP);
}
